// GATEncoder_26508538151539
// MI455X (gfx1250) — matrix operand order repaired and checked by reading, not compiled, not run on hardware
//
#include <hip/hip_runtime.h>

// ---------------- CDNA5 WMMA types ----------------
typedef __attribute__((ext_vector_type(16))) __bf16 v16bf;
typedef __attribute__((ext_vector_type(8)))  __bf16 v8bf;
typedef __attribute__((ext_vector_type(8)))  float  v8f;

#define N_NODES 20000
#define N_EDGES 200000
#define N_GRAPHS 128
#define IN_DIM 256
#define HID 512
#define HEADS 8
#define LAT 256

// ---------------- small utility kernels ----------------
__global__ void fill_f32(float* __restrict__ p, float v, size_t n) {
  size_t i = (size_t)blockIdx.x * blockDim.x + threadIdx.x;
  if (i < n) p[i] = v;
}

__global__ void cvt_bf16(const float* __restrict__ in, __bf16* __restrict__ out, size_t n) {
  size_t i = (size_t)blockIdx.x * blockDim.x + threadIdx.x;
  if (i < n) out[i] = (__bf16)in[i];
}

// W[K,N] (row-major) -> Wt[N,K] bf16 (row-major), i.e. B pre-transposed for WMMA
__global__ void tcvt_bf16(const float* __restrict__ W, __bf16* __restrict__ Wt, int K, int N) {
  size_t i = (size_t)blockIdx.x * blockDim.x + threadIdx.x;
  if (i < (size_t)K * N) {
    int nn = (int)(i / K);
    int kk = (int)(i % K);
    Wt[i] = (__bf16)W[(size_t)kk * N + nn];
  }
}

// ---------------- WMMA fragment loaders ----------------
// 16-bit A 16x32 frag, this file's k order (same as B): lane(row,half) holds K in [half*16,half*16+16) contiguous
__device__ inline v16bf load_afrag(const __bf16* __restrict__ rowp, int k0, int half) {
  v8bf lo = *(const v8bf*)(rowp + k0 + half * 16);
  v8bf hi = *(const v8bf*)(rowp + k0 + half * 16 + 8);
  v16bf a;
#pragma unroll
  for (int i = 0; i < 8; ++i) { a[i] = lo[i]; a[8 + i] = hi[i]; }
  return a;
}

// ---------------- bf16 WMMA GEMM: Y[M,N] = A[M,K] @ Bt[N,K]^T + bias ----------------
// one wave computes a 32x64 output tile (2 A-frags x 4 B-frags = 8 WMMAs per K-step)
// raises arithmetic intensity ~2.6x vs a single 16x16 tile per wave.
// requires M % 32 == 0, N % 64 == 0, K % 32 == 0 (true: M=20000, N=512, K=256/512).
__global__ void gemm_bf16_wmma(const __bf16* __restrict__ A,
                               const __bf16* __restrict__ Bt,
                               const float* __restrict__ bias,
                               float* __restrict__ Y,
                               int M, int N, int K) {
  int lane = threadIdx.x & 31;
  int waveInBlock = threadIdx.x >> 5;
  int tilesN = N >> 6;                       // 64-wide column blocks
  int totalTiles = (M >> 5) * tilesN;        // 32-tall row blocks
  int wid = blockIdx.x * (blockDim.x >> 5) + waveInBlock;
  if (wid >= totalTiles) return;
  int tm = wid / tilesN;
  int tn = wid % tilesN;

  int row  = lane & 15;
  int half = lane >> 4;

  const __bf16* arow0 = A + (size_t)(tm * 32 + row) * K;
  const __bf16* arow1 = A + (size_t)(tm * 32 + 16 + row) * K;
  const __bf16* brow0 = Bt + (size_t)(tn * 64 + row) * K;       // B 32x16: lane holds
  const __bf16* brow1 = Bt + (size_t)(tn * 64 + 16 + row) * K;  // K in [half*16, half*16+16)
  const __bf16* brow2 = Bt + (size_t)(tn * 64 + 32 + row) * K;  // contiguous per lane
  const __bf16* brow3 = Bt + (size_t)(tn * 64 + 48 + row) * K;

  v8f acc[2][4] = {};
  for (int k0 = 0; k0 < K; k0 += 32) {
    v16bf af[2];
    af[0] = load_afrag(arow0, k0, half);
    af[1] = load_afrag(arow1, k0, half);
    v16bf bf[4];
    bf[0] = *(const v16bf*)(brow0 + k0 + half * 16);
    bf[1] = *(const v16bf*)(brow1 + k0 + half * 16);
    bf[2] = *(const v16bf*)(brow2 + k0 + half * 16);
    bf[3] = *(const v16bf*)(brow3 + k0 + half * 16);
#pragma unroll
    for (int i = 0; i < 2; ++i)
#pragma unroll
      for (int j = 0; j < 4; ++j)
        acc[i][j] = __builtin_amdgcn_wmma_f32_16x16x32_bf16(false, af[i], false, bf[j],
                                                            (short)0, acc[i][j], false, false);
  }

  // C/D layout: lane -> col = lane&15, rows half*8+v
#pragma unroll
  for (int i = 0; i < 2; ++i) {
#pragma unroll
    for (int j = 0; j < 4; ++j) {
      float bv = bias[tn * 64 + j * 16 + row];
      float* yp = Y + (size_t)(tm * 32 + i * 16 + half * 8) * N + tn * 64 + j * 16 + row;
#pragma unroll
      for (int v = 0; v < 8; ++v) yp[(size_t)v * N] = acc[i][j][v] + bv;
    }
  }
}

// ---------------- edge kernels ----------------
__device__ inline void atomicMaxF(float* addr, float val) {
  if (val >= 0.f) atomicMax((int*)addr, __float_as_int(val));
  else            atomicMin((unsigned int*)addr, __float_as_uint(val));
}

// grid-stride, one wave per edge: logit[e,h] = att_h . leaky_relu(xl[src] + xr[dst])
__global__ void edge_logits(const float* __restrict__ xl, const float* __restrict__ xr,
                            const int* __restrict__ src, const int* __restrict__ dst,
                            const float* __restrict__ att,
                            float* __restrict__ P, float* __restrict__ Mx, int E) {
  int lane = threadIdx.x & 31;
  int wpb = blockDim.x >> 5;
  int gw = blockIdx.x * wpb + (threadIdx.x >> 5);
  int stride = gridDim.x * wpb;
  const float4* pa = (const float4*)att + lane * 4;
  float4 w0 = pa[0], w1 = pa[1], w2 = pa[2], w3 = pa[3];
  for (int e = gw; e < E; e += stride) {
    int s = src[e], d = dst[e];
    if (e + stride < E) {  // hide gather latency of the next edge's rows
      __builtin_prefetch(xl + (size_t)src[e + stride] * HID + lane * 16, 0, 3);
      __builtin_prefetch(xr + (size_t)dst[e + stride] * HID + lane * 16, 0, 3);
    }
    const float4* pl = (const float4*)(xl + (size_t)s * HID) + lane * 4;
    const float4* pr = (const float4*)(xr + (size_t)d * HID) + lane * 4;
    float sum = 0.f, v;
    float4 a, b;
    a = pl[0]; b = pr[0];
    v = a.x + b.x; v = v > 0.f ? v : 0.2f * v; sum += v * w0.x;
    v = a.y + b.y; v = v > 0.f ? v : 0.2f * v; sum += v * w0.y;
    v = a.z + b.z; v = v > 0.f ? v : 0.2f * v; sum += v * w0.z;
    v = a.w + b.w; v = v > 0.f ? v : 0.2f * v; sum += v * w0.w;
    a = pl[1]; b = pr[1];
    v = a.x + b.x; v = v > 0.f ? v : 0.2f * v; sum += v * w1.x;
    v = a.y + b.y; v = v > 0.f ? v : 0.2f * v; sum += v * w1.y;
    v = a.z + b.z; v = v > 0.f ? v : 0.2f * v; sum += v * w1.z;
    v = a.w + b.w; v = v > 0.f ? v : 0.2f * v; sum += v * w1.w;
    a = pl[2]; b = pr[2];
    v = a.x + b.x; v = v > 0.f ? v : 0.2f * v; sum += v * w2.x;
    v = a.y + b.y; v = v > 0.f ? v : 0.2f * v; sum += v * w2.y;
    v = a.z + b.z; v = v > 0.f ? v : 0.2f * v; sum += v * w2.z;
    v = a.w + b.w; v = v > 0.f ? v : 0.2f * v; sum += v * w2.w;
    a = pl[3]; b = pr[3];
    v = a.x + b.x; v = v > 0.f ? v : 0.2f * v; sum += v * w3.x;
    v = a.y + b.y; v = v > 0.f ? v : 0.2f * v; sum += v * w3.y;
    v = a.z + b.z; v = v > 0.f ? v : 0.2f * v; sum += v * w3.z;
    v = a.w + b.w; v = v > 0.f ? v : 0.2f * v; sum += v * w3.w;
    // reduce the 4 lanes of each head (lane l covers features [l*16,l*16+16) => head l>>2)
    sum += __shfl_xor(sum, 1, 32);
    sum += __shfl_xor(sum, 2, 32);
    if ((lane & 3) == 0) {
      int h = lane >> 2;
      P[(size_t)e * HEADS + h] = sum;
      atomicMaxF(Mx + (size_t)d * HEADS + h, sum);
    }
  }
}

// p = exp(logit - m[dst]); atomic add into denom
__global__ void edge_softmax(float* __restrict__ P, const float* __restrict__ Mx,
                             float* __restrict__ DEN, const int* __restrict__ dst, int E) {
  size_t i = (size_t)blockIdx.x * blockDim.x + threadIdx.x;
  if (i >= (size_t)E * HEADS) return;
  int e = (int)(i >> 3), h = (int)(i & 7);
  int d = dst[e];
  float p = __expf(P[i] - Mx[(size_t)d * HEADS + h]);
  P[i] = p;
  atomicAdd(DEN + (size_t)d * HEADS + h, p);
}

// grid-stride, one wave per edge: AGG[dst] += xl[src] * alpha[e,h]
__global__ void edge_message(const float* __restrict__ xl, const float* __restrict__ P,
                             const float* __restrict__ DEN,
                             const int* __restrict__ src, const int* __restrict__ dst,
                             float* __restrict__ AGG, int E) {
  int lane = threadIdx.x & 31;
  int wpb = blockDim.x >> 5;
  int gw = blockIdx.x * wpb + (threadIdx.x >> 5);
  int stride = gridDim.x * wpb;
  for (int e = gw; e < E; e += stride) {
    int s = src[e], d = dst[e];
    if (e + stride < E) {
      __builtin_prefetch(xl + (size_t)src[e + stride] * HID + lane * 16, 0, 3);
      __builtin_prefetch(AGG + (size_t)dst[e + stride] * HID + lane * 16, 1, 3);
    }
    int h = lane >> 2;
    float alpha = P[(size_t)e * HEADS + h] / (DEN[(size_t)d * HEADS + h] + 1e-16f);
    const float4* pl = (const float4*)(xl + (size_t)s * HID) + lane * 4;
    float* po = AGG + (size_t)d * HID + lane * 16;
#pragma unroll
    for (int j = 0; j < 4; ++j) {
      float4 a = pl[j];
      atomicAdd(po + j * 4 + 0, a.x * alpha);
      atomicAdd(po + j * 4 + 1, a.y * alpha);
      atomicAdd(po + j * 4 + 2, a.z * alpha);
      atomicAdd(po + j * 4 + 3, a.w * alpha);
    }
  }
}

__global__ void bias_elu(float* __restrict__ H, const float* __restrict__ bias, size_t n) {
  size_t i = (size_t)blockIdx.x * blockDim.x + threadIdx.x;
  if (i < n) {
    float v = H[i] + bias[i & (HID - 1)];
    H[i] = v > 0.f ? v : (__expf(v) - 1.f);
  }
}

__global__ void pool_add(const float* __restrict__ H, const int* __restrict__ batch,
                         float* __restrict__ G, size_t n) {
  size_t i = (size_t)blockIdx.x * blockDim.x + threadIdx.x;
  if (i < n) {
    int node = (int)(i >> 9);
    atomicAdd(G + (size_t)batch[node] * HID + (i & (HID - 1)), H[i]);
  }
}

__global__ void bn_kernel(const float* __restrict__ G, const float* __restrict__ gamma,
                          const float* __restrict__ beta, const float* __restrict__ mean,
                          const float* __restrict__ var, float* __restrict__ GBN, int n) {
  int i = blockIdx.x * blockDim.x + threadIdx.x;
  if (i < n) {
    int f = i & (HID - 1);
    GBN[i] = (G[i] - mean[f]) * rsqrtf(var[f] + 1e-5f) * gamma[f] + beta[f];
  }
}

__global__ void fc_kernel(const float* __restrict__ GBN, const float* __restrict__ W,
                          const float* __restrict__ b, float* __restrict__ out) {
  int i = blockIdx.x * blockDim.x + threadIdx.x;
  if (i >= N_GRAPHS * LAT) return;
  int g = i >> 8, o = i & (LAT - 1);
  const float* gr = GBN + (size_t)g * HID;
  float s = b[o];
#pragma unroll 8
  for (int k = 0; k < HID; ++k) s += gr[k] * W[(size_t)k * LAT + o];
  out[i] = s;
}

// ---------------- launch ----------------
static inline int cdiv(long long a, int b) { return (int)((a + b - 1) / b); }

extern "C" void kernel_launch(void* const* d_in, const int* in_sizes, int n_in,
                              void* d_out, int out_size, void* d_ws, size_t ws_size,
                              hipStream_t stream) {
  const float* x    = (const float*)d_in[0];
  const int*   ei   = (const int*)d_in[1];
  const int*   src  = ei;
  const int*   dst  = ei + N_EDGES;
  const int*   batch= (const int*)d_in[2];
  const float *Wl0=(const float*)d_in[3],  *bl0=(const float*)d_in[4];
  const float *Wr0=(const float*)d_in[5],  *br0=(const float*)d_in[6];
  const float *att0=(const float*)d_in[7], *bias0=(const float*)d_in[8];
  const float *Wl1=(const float*)d_in[9],  *bl1=(const float*)d_in[10];
  const float *Wr1=(const float*)d_in[11], *br1=(const float*)d_in[12];
  const float *att1=(const float*)d_in[13],*bias1=(const float*)d_in[14];
  const float *gamma=(const float*)d_in[15],*beta=(const float*)d_in[16];
  const float *mean=(const float*)d_in[17], *var=(const float*)d_in[18];
  const float *Wfc=(const float*)d_in[19],  *bfc=(const float*)d_in[20];
  float* out = (float*)d_out;

  // workspace carving (256-byte aligned slabs, ~111 MB total)
  size_t off = 0;
  auto take = [&](size_t bytes) { size_t r = off; off += (bytes + 255) & ~(size_t)255; return r; };
  char* ws = (char*)d_ws;
  __bf16* BFX = (__bf16*)(ws + take((size_t)N_NODES * HID * 2));   // bf16 activations (K<=512)
  __bf16* WTL = (__bf16*)(ws + take((size_t)HID * HID * 2));       // transposed bf16 Wl
  __bf16* WTR = (__bf16*)(ws + take((size_t)HID * HID * 2));       // transposed bf16 Wr
  float*  B0  = (float*)(ws + take((size_t)N_NODES * HID * 4));    // xl
  float*  B1  = (float*)(ws + take((size_t)N_NODES * HID * 4));    // xr -> agg -> h
  float*  P   = (float*)(ws + take((size_t)N_EDGES * HEADS * 4));  // logits -> probs
  float*  Mx  = (float*)(ws + take((size_t)N_NODES * HEADS * 4));  // segment max
  float*  DEN = (float*)(ws + take((size_t)N_NODES * HEADS * 4));  // segment sum
  float*  G   = (float*)(ws + take((size_t)N_GRAPHS * HID * 4));   // pooled
  float*  GBN = (float*)(ws + take((size_t)N_GRAPHS * HID * 4));   // batch-normed

  const int T = 256;
  size_t nNH = (size_t)N_NODES * HID;
  size_t nNHd= (size_t)N_NODES * HEADS;
  size_t nEH = (size_t)N_EDGES * HEADS;
  int gemmBlocks = cdiv((long long)(N_NODES / 32) * (HID / 64), 8); // 32x64 tile per wave, 8 waves/block
  const int EGRID = 1024;                                           // grid-stride edge kernels

  // ---------- layer 0 ----------
  cvt_bf16<<<cdiv((long long)N_NODES * IN_DIM, T), T, 0, stream>>>(x, BFX, (size_t)N_NODES * IN_DIM);
  tcvt_bf16<<<cdiv((long long)IN_DIM * HID, T), T, 0, stream>>>(Wl0, WTL, IN_DIM, HID);
  tcvt_bf16<<<cdiv((long long)IN_DIM * HID, T), T, 0, stream>>>(Wr0, WTR, IN_DIM, HID);
  fill_f32<<<cdiv((long long)nNHd, T), T, 0, stream>>>(Mx, -INFINITY, nNHd);
  fill_f32<<<cdiv((long long)nNHd, T), T, 0, stream>>>(DEN, 0.f, nNHd);
  gemm_bf16_wmma<<<gemmBlocks, T, 0, stream>>>(BFX, WTL, bl0, B0, N_NODES, HID, IN_DIM);
  gemm_bf16_wmma<<<gemmBlocks, T, 0, stream>>>(BFX, WTR, br0, B1, N_NODES, HID, IN_DIM);
  edge_logits<<<EGRID, T, 0, stream>>>(B0, B1, src, dst, att0, P, Mx, N_EDGES);
  edge_softmax<<<cdiv((long long)nEH, T), T, 0, stream>>>(P, Mx, DEN, dst, N_EDGES);
  fill_f32<<<cdiv((long long)nNH, T), T, 0, stream>>>(B1, 0.f, nNH);       // XR dead -> becomes AGG
  edge_message<<<EGRID, T, 0, stream>>>(B0, P, DEN, src, dst, B1, N_EDGES);
  bias_elu<<<cdiv((long long)nNH, T), T, 0, stream>>>(B1, bias0, nNH);     // h0 in B1

  // ---------- layer 1 ----------
  cvt_bf16<<<cdiv((long long)nNH, T), T, 0, stream>>>(B1, BFX, nNH);
  tcvt_bf16<<<cdiv((long long)HID * HID, T), T, 0, stream>>>(Wl1, WTL, HID, HID);
  tcvt_bf16<<<cdiv((long long)HID * HID, T), T, 0, stream>>>(Wr1, WTR, HID, HID);
  fill_f32<<<cdiv((long long)nNHd, T), T, 0, stream>>>(Mx, -INFINITY, nNHd);
  fill_f32<<<cdiv((long long)nNHd, T), T, 0, stream>>>(DEN, 0.f, nNHd);
  gemm_bf16_wmma<<<gemmBlocks, T, 0, stream>>>(BFX, WTL, bl1, B0, N_NODES, HID, HID);
  gemm_bf16_wmma<<<gemmBlocks, T, 0, stream>>>(BFX, WTR, br1, B1, N_NODES, HID, HID);
  edge_logits<<<EGRID, T, 0, stream>>>(B0, B1, src, dst, att1, P, Mx, N_EDGES);
  edge_softmax<<<cdiv((long long)nEH, T), T, 0, stream>>>(P, Mx, DEN, dst, N_EDGES);
  fill_f32<<<cdiv((long long)nNH, T), T, 0, stream>>>(B1, 0.f, nNH);
  edge_message<<<EGRID, T, 0, stream>>>(B0, P, DEN, src, dst, B1, N_EDGES);
  bias_elu<<<cdiv((long long)nNH, T), T, 0, stream>>>(B1, bias1, nNH);     // h1 in B1

  // ---------- pool + BN + FC ----------
  fill_f32<<<cdiv((long long)N_GRAPHS * HID, T), T, 0, stream>>>(G, 0.f, (size_t)N_GRAPHS * HID);
  pool_add<<<cdiv((long long)nNH, T), T, 0, stream>>>(B1, batch, G, nNH);
  bn_kernel<<<cdiv((long long)N_GRAPHS * HID, T), T, 0, stream>>>(G, gamma, beta, mean, var, GBN, N_GRAPHS * HID);
  fc_kernel<<<cdiv((long long)N_GRAPHS * LAT, T), T, 0, stream>>>(GBN, Wfc, bfc, out);
}
